// MambaBlock_18545668784335
// MI455X (gfx1250) — matrix operand order repaired and checked by reading, not compiled, not run on hardware
//
#include <hip/hip_runtime.h>

#define D_MODEL 1024
#define D_STATE 16
#define D_CONV  4
#define D_INNER 2048
#define BATCH   2
#define SEQ     2048
#define NTOK    (BATCH * SEQ)            // 4096 tokens
#define DXZ     (2 * D_INNER)            // 4096
#define DBCD    (2 * D_STATE + D_INNER)  // 2080
#define LN_EPS  1e-5f
#define NCHUNK  32
#define CHUNK   (SEQ / NCHUNK)           // 64

typedef __attribute__((ext_vector_type(16))) _Float16 v16h;
typedef __attribute__((ext_vector_type(8)))  _Float16 v8h;
typedef __attribute__((ext_vector_type(8)))  float    v8f;

// ---------------- LayerNorm over D_MODEL per token; emit f16 activations ----
__global__ void ln_kernel(const float* __restrict__ x,
                          const float* __restrict__ w,
                          const float* __restrict__ b,
                          _Float16* __restrict__ xn) {
  __shared__ float red[256];
  const int t  = blockIdx.x;          // token = b*SEQ + l
  const int bi = t / SEQ, l = t % SEQ;
  const float* xb = x + (size_t)bi * D_MODEL * SEQ + l;   // + c*SEQ strided
  float s = 0.f;
  for (int c = threadIdx.x; c < D_MODEL; c += 256) s += xb[(size_t)c * SEQ];
  red[threadIdx.x] = s; __syncthreads();
  for (int o = 128; o > 0; o >>= 1) {
    if (threadIdx.x < o) red[threadIdx.x] += red[threadIdx.x + o];
    __syncthreads();
  }
  const float mu = red[0] * (1.f / D_MODEL);
  __syncthreads();
  float v = 0.f;
  for (int c = threadIdx.x; c < D_MODEL; c += 256) {
    float d = xb[(size_t)c * SEQ] - mu; v += d * d;
  }
  red[threadIdx.x] = v; __syncthreads();
  for (int o = 128; o > 0; o >>= 1) {
    if (threadIdx.x < o) red[threadIdx.x] += red[threadIdx.x + o];
    __syncthreads();
  }
  const float rstd = rsqrtf(red[0] * (1.f / D_MODEL) + LN_EPS);
  for (int c = threadIdx.x; c < D_MODEL; c += 256)
    xn[(size_t)t * D_MODEL + c] =
        (_Float16)((xb[(size_t)c * SEQ] - mu) * rstd * w[c] + b[c]);
}

// ------------- f32 [K,N] -> f16 transposed [N,K] (weights, once per launch) -
__global__ void transpose_f16_kernel(const float* __restrict__ W,
                                     _Float16* __restrict__ WT,
                                     int K, int N) {
  size_t i = (size_t)blockIdx.x * blockDim.x + threadIdx.x;
  if (i >= (size_t)K * N) return;
  int k = (int)(i / N), n = (int)(i % N);
  WT[(size_t)n * K + k] = (_Float16)W[i];
}

// ---------------- WMMA f16 GEMM: C[M,N] = A[M,K] * BT[N,K]^T ---------------
// Register-blocked 32x64 macro tile per wave (2 M-tiles x 4 N-tiles => 8
// accumulators).  K and N are template parameters so every inter-tile stride
// (j*16*K, mi*16*K, k0, r*N) is a compile-time constant folded into the
// 24-bit global memory immediate offset: two load base pairs + one store
// base pair per tile, ~120 VGPRs, no spills, no address-add chains.
// BT must be padded to a multiple of 4 N-tiles (64 rows); stores are guarded
// (guard folds away when N % 64 == 0).
template <int K, int N>
__global__ void gemm_wmma_kernel(const _Float16* __restrict__ A,
                                 const _Float16* __restrict__ BT,
                                 float* __restrict__ C) {
  constexpr int nTilesN = (N + 15) / 16;
  constexpr int nMacroN = (nTilesN + 3) / 4;
  const int wave = threadIdx.x >> 5;
  const int lane = threadIdx.x & 31;
  const int grp  = lane >> 4;     // 0 or 1 (lane half)
  const int idx  = lane & 15;     // row within A-tile / row within B^T tile
  const int macroN  = blockIdx.x * 4 + wave;   // wave-uniform guard
  const int macroM  = blockIdx.y;              // 32 rows
  if (macroN >= nMacroN) return;
  const int tileN0 = macroN * 4;

  const _Float16* Ap = A  + ((size_t)macroM * 32 + idx) * K;
  const _Float16* Bp = BT + ((size_t)tileN0 * 16 + idx) * K;

  v8f acc[2][4] = {};
  for (int k0 = 0; k0 < K; k0 += 32) {
    __builtin_prefetch(Ap + k0 + 1024, 0, 1);
    __builtin_prefetch(Bp + k0 + 1024, 0, 1);
    // A 16x32 f16 frag: lane half grp covers K = k0+grp*16 .. +15 (contiguous)
    v16h af[2];
#pragma unroll
    for (int mi = 0; mi < 2; ++mi) {
      af[mi] = *(const v16h*)(Ap + mi * 16 * K + k0 + grp * 16);
    }
    // B 32x16 f16 frag: lane half grp covers K = k0+grp*16 .. +15 (contiguous)
    v16h bf[4];
#pragma unroll
    for (int j = 0; j < 4; ++j)
      bf[j] = *(const v16h*)(Bp + j * 16 * K + k0 + grp * 16);
#pragma unroll
    for (int mi = 0; mi < 2; ++mi)
#pragma unroll
      for (int j = 0; j < 4; ++j)
        acc[mi][j] = __builtin_amdgcn_wmma_f32_16x16x32_f16(
            false, af[mi], false, bf[j], (short)0, acc[mi][j], false, false);
  }
  // C/D layout: VGPR r -> row m0+r, col n (per lane); r*N folds to immediates
#pragma unroll
  for (int mi = 0; mi < 2; ++mi) {
    const int m0 = macroM * 32 + mi * 16 + grp * 8;
#pragma unroll
    for (int j = 0; j < 4; ++j) {
      const int tn = tileN0 + j;
      if (nTilesN % 4 == 0 || tn < nTilesN) {      // folds when N%64==0
        float* Crow = C + (size_t)m0 * N + tn * 16 + idx;
#pragma unroll
        for (int r = 0; r < 8; ++r)
          Crow[r * N] = acc[mi][j][r];
      }
    }
  }
}

// ------------- causal depthwise conv (k=4) + bias + SiLU -------------------
__global__ void conv_silu_kernel(const float* __restrict__ xz,
                                 const float* __restrict__ cw,
                                 const float* __restrict__ cb,
                                 _Float16* __restrict__ xh,
                                 float* __restrict__ xf) {
  size_t i = (size_t)blockIdx.x * blockDim.x + threadIdx.x;
  if (i >= (size_t)NTOK * D_INNER) return;
  const int d = (int)(i % D_INNER);
  const int t = (int)(i / D_INNER);
  const int l = t % SEQ, b = t / SEQ;
  float acc = cb[d];
#pragma unroll
  for (int j = 0; j < D_CONV; ++j) {
    int li = l - (D_CONV - 1) + j;
    if (li >= 0)
      acc += cw[d * D_CONV + j] * xz[((size_t)(b * SEQ + li)) * DXZ + d];
  }
  acc = acc / (1.f + __expf(-acc));   // SiLU
  xh[i] = (_Float16)acc;
  xf[i] = acc;
}

// ------------- extract dt-projection slice of bcd to f16 -------------------
__global__ void dproj_kernel(const float* __restrict__ bcd,
                             _Float16* __restrict__ dh) {
  size_t i = (size_t)blockIdx.x * blockDim.x + threadIdx.x;
  if (i >= (size_t)NTOK * D_INNER) return;
  const int t = (int)(i / D_INNER), c = (int)(i % D_INNER);
  dh[i] = (_Float16)bcd[(size_t)t * DBCD + 2 * D_STATE + c];
}

// ------------- dt = softplus(dtlin + b_dt), in place -----------------------
__global__ void softplus_kernel(float* __restrict__ dt,
                                const float* __restrict__ bdt) {
  size_t i = (size_t)blockIdx.x * blockDim.x + threadIdx.x;
  if (i >= (size_t)NTOK * D_INNER) return;
  const int c = (int)(i % D_INNER);
  float v = dt[i] + bdt[c];
  dt[i] = (v > 20.f) ? v : log1pf(__expf(v));
}

// ===================== chunked selective scan ==============================
// h_t = exp(dlt_t*A)*h_{t-1} + dlt_t*B_t*u_t  is associative per state, and
// the chunk decay product collapses: prod_t exp(dlt_t*A_s) = exp(A_s*sum dlt).
// Phase 1: per (b,chunk,d) compute chunk-local end state (from h=0) + sum(dlt).
// Phase 2: per (b,d) 32-step prefix combine -> h_init per chunk.
// Phase 3: per (b,chunk,d) replay chunk from h_init, emit y.
// ===========================================================================
__global__ void scan_phase1_kernel(const float* __restrict__ dt,
                                   const float* __restrict__ bcd,
                                   const float* __restrict__ u,
                                   const float* __restrict__ A_log,
                                   float* __restrict__ chunkH,   // [B*NCHUNK*D_INNER, 16]
                                   float* __restrict__ sumdlt) { // [B*NCHUNK*D_INNER]
  const int gid = blockIdx.x * blockDim.x + threadIdx.x;
  if (gid >= BATCH * NCHUNK * D_INNER) return;
  const int d     = gid % D_INNER;
  const int rem   = gid / D_INNER;
  const int chunk = rem % NCHUNK;
  const int b     = rem / NCHUNK;
  float A[D_STATE], h[D_STATE];
#pragma unroll
  for (int s = 0; s < D_STATE; ++s) {
    A[s] = -__expf(A_log[d * D_STATE + s]);
    h[s] = 0.f;
  }
  float sd = 0.f;
  for (int l = chunk * CHUNK; l < chunk * CHUNK + CHUNK; ++l) {
    const size_t t = (size_t)b * SEQ + l;
    const float dlt = dt[t * D_INNER + d];
    const float uu  = u[t * D_INNER + d];
    const float* bc = bcd + t * DBCD;
    const float du = dlt * uu;
    sd += dlt;
#pragma unroll
    for (int s = 0; s < D_STATE; ++s)
      h[s] = __expf(dlt * A[s]) * h[s] + du * bc[s];
  }
  const size_t base = (size_t)gid * D_STATE;
#pragma unroll
  for (int s = 0; s < D_STATE; ++s) chunkH[base + s] = h[s];
  sumdlt[gid] = sd;
}

__global__ void scan_phase2_kernel(const float* __restrict__ chunkH,
                                   const float* __restrict__ sumdlt,
                                   const float* __restrict__ A_log,
                                   float* __restrict__ hinit) {  // [B*NCHUNK*D_INNER, 16]
  const int gid = blockIdx.x * blockDim.x + threadIdx.x;
  if (gid >= BATCH * D_INNER) return;
  const int b = gid / D_INNER;
  const int d = gid % D_INNER;
  float A[D_STATE], h[D_STATE];
#pragma unroll
  for (int s = 0; s < D_STATE; ++s) {
    A[s] = -__expf(A_log[d * D_STATE + s]);
    h[s] = 0.f;
  }
  for (int c = 0; c < NCHUNK; ++c) {
    const size_t idx = (size_t)(b * NCHUNK + c) * D_INNER + d;
#pragma unroll
    for (int s = 0; s < D_STATE; ++s) hinit[idx * D_STATE + s] = h[s];
    const float dA = sumdlt[idx];
#pragma unroll
    for (int s = 0; s < D_STATE; ++s)
      h[s] = __expf(dA * A[s]) * h[s] + chunkH[idx * D_STATE + s];
  }
}

__global__ void scan_phase3_kernel(const float* __restrict__ dt,
                                   const float* __restrict__ bcd,
                                   const float* __restrict__ u,
                                   const float* __restrict__ A_log,
                                   const float* __restrict__ Dskip,
                                   const float* __restrict__ hinit,
                                   float* __restrict__ y) {
  const int gid = blockIdx.x * blockDim.x + threadIdx.x;
  if (gid >= BATCH * NCHUNK * D_INNER) return;
  const int d     = gid % D_INNER;
  const int rem   = gid / D_INNER;
  const int chunk = rem % NCHUNK;
  const int b     = rem / NCHUNK;
  float A[D_STATE], h[D_STATE];
  const size_t base = (size_t)gid * D_STATE;
#pragma unroll
  for (int s = 0; s < D_STATE; ++s) {
    A[s] = -__expf(A_log[d * D_STATE + s]);
    h[s] = hinit[base + s];
  }
  const float Dd = Dskip[d];
  for (int l = chunk * CHUNK; l < chunk * CHUNK + CHUNK; ++l) {
    const size_t t = (size_t)b * SEQ + l;
    const float dlt = dt[t * D_INNER + d];
    const float uu  = u[t * D_INNER + d];
    const float* bc = bcd + t * DBCD;
    const float du = dlt * uu;
    float acc = 0.f;
#pragma unroll
    for (int s = 0; s < D_STATE; ++s) {
      h[s] = __expf(dlt * A[s]) * h[s] + du * bc[s];
      acc += h[s] * bc[D_STATE + s];
    }
    y[t * D_INNER + d] = acc + Dd * uu;
  }
}

// ------------- gate: yh = (f16)(y * silu(z)) -------------------------------
__global__ void gate_kernel(const float* __restrict__ y,
                            const float* __restrict__ xz,
                            _Float16* __restrict__ yh) {
  size_t i = (size_t)blockIdx.x * blockDim.x + threadIdx.x;
  if (i >= (size_t)NTOK * D_INNER) return;
  const int t = (int)(i / D_INNER), d = (int)(i % D_INNER);
  const float z = xz[(size_t)t * DXZ + D_INNER + d];
  const float g = z / (1.f + __expf(-z));
  yh[i] = (_Float16)(y[i] * g);
}

// ------------- residual add + (B,L,D)->(B,D,L) transpose -------------------
__global__ void out_kernel(const float* __restrict__ og,   // [NTOK, D_MODEL]
                           const float* __restrict__ x,    // [B, D_MODEL, SEQ]
                           float* __restrict__ out) {
  size_t i = (size_t)blockIdx.x * blockDim.x + threadIdx.x;
  if (i >= (size_t)BATCH * D_MODEL * SEQ) return;
  const int l = (int)(i % SEQ);
  const size_t r = i / SEQ;
  const int c = (int)(r % D_MODEL);
  const int b = (int)(r / D_MODEL);
  out[i] = og[((size_t)(b * SEQ + l)) * D_MODEL + c] + x[i];
}

extern "C" void kernel_launch(void* const* d_in, const int* in_sizes, int n_in,
                              void* d_out, int out_size, void* d_ws, size_t ws_size,
                              hipStream_t stream) {
  (void)in_sizes; (void)n_in; (void)out_size; (void)ws_size;
  const float* x      = (const float*)d_in[0];
  const float* ln_w   = (const float*)d_in[1];
  const float* ln_b   = (const float*)d_in[2];
  const float* W_in   = (const float*)d_in[3];   // [1024, 4096]
  const float* conv_w = (const float*)d_in[4];   // [2048, 1, 4]
  const float* conv_b = (const float*)d_in[5];
  const float* W_x    = (const float*)d_in[6];   // [2048, 2080]
  const float* W_dt   = (const float*)d_in[7];   // [2048, 2048]
  const float* b_dt   = (const float*)d_in[8];
  const float* A_log  = (const float*)d_in[9];
  const float* Dskip  = (const float*)d_in[10];
  const float* W_out  = (const float*)d_in[11];  // [2048, 1024]
  float* out = (float*)d_out;

  // pad transposed weights to a multiple of 64 rows (4 N-tiles) so the GEMM
  // can load ragged-N B-frags unguarded (stores are guarded)
  auto padN = [](int N) { return (size_t)((((N + 15) / 16 + 3) & ~3) * 16); };

  char* ws = (char*)d_ws;
  size_t off = 0;
  auto alloc = [&](size_t bytes) -> char* {
    char* p = ws + off;
    off += (bytes + 255) & ~(size_t)255;
    return p;
  };
  _Float16* xn_h  = (_Float16*)alloc((size_t)NTOK * D_MODEL * 2);
  _Float16* WinT  = (_Float16*)alloc(padN(DXZ) * D_MODEL * 2);
  float*    xz    = (float*)   alloc((size_t)NTOK * DXZ * 4);
  _Float16* xb_h  = (_Float16*)alloc((size_t)NTOK * D_INNER * 2);
  float*    xb_f  = (float*)   alloc((size_t)NTOK * D_INNER * 4);
  _Float16* WxT   = (_Float16*)alloc(padN(DBCD) * D_INNER * 2);
  float*    bcd   = (float*)   alloc((size_t)NTOK * DBCD * 4);
  _Float16* dp_h  = (_Float16*)alloc((size_t)NTOK * D_INNER * 2);
  _Float16* WdtT  = (_Float16*)alloc(padN(D_INNER) * D_INNER * 2);
  float*    dt    = (float*)   alloc((size_t)NTOK * D_INNER * 4);
  float*    yb    = (float*)   alloc((size_t)NTOK * D_INNER * 4);
  _Float16* y_h   = (_Float16*)alloc((size_t)NTOK * D_INNER * 2);
  _Float16* WoutT = (_Float16*)alloc(padN(D_MODEL) * D_INNER * 2);
  float*    og    = (float*)   alloc((size_t)NTOK * D_MODEL * 4);
  float*    chunkH = (float*)  alloc((size_t)BATCH * NCHUNK * D_INNER * D_STATE * 4);
  float*    sumdlt = (float*)  alloc((size_t)BATCH * NCHUNK * D_INNER * 4);
  float*    hinit  = (float*)  alloc((size_t)BATCH * NCHUNK * D_INNER * D_STATE * 4);

  const int TPB = 256;
  auto blk = [](size_t n, int t) { return (unsigned)((n + t - 1) / t); };
  auto gemm_grid = [](int M, int N) {
    int nTilesN = (N + 15) / 16;
    int nMacroN = (nTilesN + 3) / 4;
    return dim3((unsigned)((nMacroN + 3) / 4), (unsigned)(M / 32));
  };

  // 1) LayerNorm -> f16 activations
  ln_kernel<<<NTOK, TPB, 0, stream>>>(x, ln_w, ln_b, xn_h);

  // weight down-convert + transpose (once per launch)
  transpose_f16_kernel<<<blk((size_t)D_MODEL * DXZ, TPB), TPB, 0, stream>>>(W_in, WinT, D_MODEL, DXZ);
  transpose_f16_kernel<<<blk((size_t)D_INNER * DBCD, TPB), TPB, 0, stream>>>(W_x, WxT, D_INNER, DBCD);
  transpose_f16_kernel<<<blk((size_t)D_INNER * D_INNER, TPB), TPB, 0, stream>>>(W_dt, WdtT, D_INNER, D_INNER);
  transpose_f16_kernel<<<blk((size_t)D_INNER * D_MODEL, TPB), TPB, 0, stream>>>(W_out, WoutT, D_INNER, D_MODEL);

  // 2) xz = xn @ W_in   [4096 x 4096], K=1024
  gemm_wmma_kernel<D_MODEL, DXZ><<<gemm_grid(NTOK, DXZ), 128, 0, stream>>>(xn_h, WinT, xz);

  // 3) causal depthwise conv + SiLU
  conv_silu_kernel<<<blk((size_t)NTOK * D_INNER, TPB), TPB, 0, stream>>>(xz, conv_w, conv_b, xb_h, xb_f);

  // 4) bcd = xb @ W_x   [4096 x 2080], K=2048
  gemm_wmma_kernel<D_INNER, DBCD><<<gemm_grid(NTOK, DBCD), 128, 0, stream>>>(xb_h, WxT, bcd);

  // 5) dt = softplus(dproj @ W_dt + b_dt), K=2048
  dproj_kernel<<<blk((size_t)NTOK * D_INNER, TPB), TPB, 0, stream>>>(bcd, dp_h);
  gemm_wmma_kernel<D_INNER, D_INNER><<<gemm_grid(NTOK, D_INNER), 128, 0, stream>>>(dp_h, WdtT, dt);
  softplus_kernel<<<blk((size_t)NTOK * D_INNER, TPB), TPB, 0, stream>>>(dt, b_dt);

  // 6) chunked selective scan
  scan_phase1_kernel<<<blk((size_t)BATCH * NCHUNK * D_INNER, TPB), TPB, 0, stream>>>(
      dt, bcd, xb_f, A_log, chunkH, sumdlt);
  scan_phase2_kernel<<<blk((size_t)BATCH * D_INNER, TPB), TPB, 0, stream>>>(
      chunkH, sumdlt, A_log, hinit);
  scan_phase3_kernel<<<blk((size_t)BATCH * NCHUNK * D_INNER, TPB), TPB, 0, stream>>>(
      dt, bcd, xb_f, A_log, Dskip, hinit, yb);

  // 7) gate with silu(z), then out-proj + residual, K=2048
  gate_kernel<<<blk((size_t)NTOK * D_INNER, TPB), TPB, 0, stream>>>(yb, xz, y_h);
  gemm_wmma_kernel<D_INNER, D_MODEL><<<gemm_grid(NTOK, D_MODEL), 128, 0, stream>>>(y_h, WoutT, og);
  out_kernel<<<blk((size_t)BATCH * D_MODEL * SEQ, TPB), TPB, 0, stream>>>(og, x, out);
}
